// LSTM_32890859553340
// MI455X (gfx1250) — matrix operand order repaired and checked by reading, not compiled, not run on hardware
//
#include <hip/hip_runtime.h>
#include <math.h>

#define B_    256
#define T_    512
#define E_    300
#define H_    128
#define FOURH 512
#define KPAD  320           // E padded to multiple of 32
#define KCX   (KPAD / 32)   // 10 K-chunks for input GEMM
#define KCH   (H_ / 32)     // 4  K-chunks for recurrent GEMM

typedef __attribute__((ext_vector_type(16))) _Float16 v16h;
typedef __attribute__((ext_vector_type(8)))  float    v8f;

__device__ __forceinline__ float sigmoidf_(float x) { return 1.0f / (1.0f + __expf(-x)); }

// ---------------------------------------------------------------------------
// Kernel 0: build fragment-permuted f16 weights.
//   WxP: B-fragment layout of W[0:300] zero-padded to K=320:  [32 ct][10 kc][32 lane][16 i]
//   WhP: B-fragment layout of W[300:428]:                     [32 ct][ 4 kc][32 lane][16 i]
//   element (ct,kc,lane,i) = W[k = kc*32 + (i&7) + 8*(lane>>4) + 16*(i>>3)][n = ct*16 + (lane&15)]
// ---------------------------------------------------------------------------
__global__ __launch_bounds__(256) void lstm_wprep(const float* __restrict__ W,
                                                  _Float16* __restrict__ WxP,
                                                  _Float16* __restrict__ WhP) {
  const int totalX = 32 * KCX * 32 * 16;   // 163840
  const int totalH = 32 * KCH * 32 * 16;   // 65536
  int idx = blockIdx.x * 256 + threadIdx.x;
  if (idx < totalX) {
    int i = idx & 15, lane = (idx >> 4) & 31, rem = idx >> 9;
    int kc = rem % KCX, ct = rem / KCX;
    int k = kc * 32 + (i & 7) + 8 * (lane >> 4) + 16 * (i >> 3);
    int n = ct * 16 + (lane & 15);
    WxP[idx] = (_Float16)((k < E_) ? W[(size_t)k * FOURH + n] : 0.0f);
  } else {
    int j = idx - totalX;
    if (j < totalH) {
      int i = j & 15, lane = (j >> 4) & 31, rem = j >> 9;
      int kc = rem & 3, ct = rem >> 2;
      int k = kc * 32 + (i & 7) + 8 * (lane >> 4) + 16 * (i >> 3);
      int n = ct * 16 + (lane & 15);
      WhP[j] = (_Float16)W[(size_t)(E_ + k) * FOURH + n];
    }
  }
}

// ---------------------------------------------------------------------------
// Kernel 1: xz[r, :] = embed[x[r], :] @ Wx  for r in [0, B*T).
// 64x64 tile per workgroup. Gathered A tile staged in LDS in A-fragment order
// (contiguous 32B per lane per fragment); B fragments are contiguous 32B
// coalesced global loads from L2-resident WxP. Result staged and stored as
// coalesced 32B chunks.
// ---------------------------------------------------------------------------
__global__ __launch_bounds__(256) void lstm_xgemm(const int* __restrict__ x,
                                                  const float* __restrict__ embed,
                                                  const _Float16* __restrict__ WxP,
                                                  _Float16* __restrict__ xz) {
  __shared__ int      sTok[64];
  __shared__ _Float16 sAP[64 * KPAD];   // 40 KB, A-fragment permuted
  __shared__ _Float16 sOut[64 * 64];    // 8 KB

  const int tid   = threadIdx.x;
  const int rbase = blockIdx.x * 64;
  const int cbase = blockIdx.y * 64;

  if (tid < 64) sTok[tid] = x[rbase + tid];
  __syncthreads();

  for (int idx = tid; idx < 64 * KPAD; idx += 256) {
    int row = idx / KPAD, col = idx - row * KPAD;
    float v = (col < E_) ? embed[(size_t)sTok[row] * E_ + col] : 0.0f;
    // scatter into A-fragment layout: k = (i&7) + 8g + 16*(i>>3)
    int mt = row >> 4, mm = row & 15, kc = col >> 5, kk = col & 31;
    int gg = (kk >> 3) & 1;
    int fi = (kk & 7) | (((kk >> 4) & 1) << 3);
    sAP[((mt * KCX + kc) * 32 + (mm + 16 * gg)) * 16 + fi] = (_Float16)v;
  }
  __syncthreads();

  const int lane = tid & 31;           // wave32
  const int w    = tid >> 5;
  const int mt   = w >> 1;             // 16-row tile 0..3
  const int nh   = (w & 1) * 32;       // 32-col half
  const int m    = lane & 15;
  const int g    = lane >> 4;
  const int ct0  = (cbase + nh) >> 4;  // first 16-col tile index

  v8f acc0 = {}; v8f acc1 = {};
  #pragma unroll
  for (int kc = 0; kc < KCX; ++kc) {
    v16h a  = *(const v16h*)&sAP[((mt * KCX + kc) * 32 + lane) * 16];
    v16h b0 = *(const v16h*)&WxP[(((size_t)ct0 * KCX + kc) * 32 + lane) * 16];
    v16h b1 = *(const v16h*)&WxP[((((size_t)ct0 + 1) * KCX + kc) * 32 + lane) * 16];
    acc0 = __builtin_amdgcn_wmma_f32_16x16x32_f16(false, a, false, b0, (short)0, acc0, false, false);
    acc1 = __builtin_amdgcn_wmma_f32_16x16x32_f16(false, a, false, b1, (short)0, acc1, false, false);
  }
  #pragma unroll
  for (int r = 0; r < 8; ++r) {        // C/D: M = r + 8g, N = lane&15
    int rl = mt * 16 + r + 8 * g;
    sOut[rl * 64 + nh + m]      = (_Float16)acc0[r];
    sOut[rl * 64 + nh + m + 16] = (_Float16)acc1[r];
  }
  __syncthreads();

  { // coalesced store: 256 threads x 32 bytes
    int row = tid >> 2, seg = tid & 3;
    v16h vdat = *(const v16h*)&sOut[row * 64 + seg * 16];
    *(v16h*)&xz[(size_t)(rbase + row) * FOURH + cbase + seg * 16] = vdat;
  }
}

// ---------------------------------------------------------------------------
// Kernel 2: recurrent scan; one workgroup per 16 batch rows.
// Per step: async-copy xz tile global->LDS (ASYNCcnt) overlapped with
// h @ Wh WMMAs (h kept in A-fragment-permuted LDS, Wh fragments are 32B
// coalesced global loads), then layernorms + gating, coalesced output.
// ---------------------------------------------------------------------------
__global__ __launch_bounds__(256) void lstm_scan(const _Float16* __restrict__ WhP,
                                                 const _Float16* __restrict__ xz,
                                                 const float* __restrict__ gains,
                                                 const float* __restrict__ shifts,
                                                 float* __restrict__ out) {
  __shared__ _Float16 sHP[KCH * 32 * 16];  // 4 KB, h in A-fragment layout
  __shared__ _Float16 sX[16 * FOURH];      // 16 KB, async-landed xz tile
  __shared__ float    sC[16 * H_];         // 8 KB
  __shared__ float    sZ[16 * FOURH];      // 32 KB
  __shared__ float    sMean[16][4], sRstd[16][4];
  __shared__ float    sCm[16], sCr[16];

  const int tid   = threadIdx.x;
  const int bbase = blockIdx.x * 16;

  for (int i2 = tid; i2 < KCH * 32 * 16; i2 += 256) sHP[i2] = (_Float16)0.0f;
  for (int i2 = tid; i2 < 16 * H_; i2 += 256) sC[i2] = 0.0f;
  __syncthreads();

  const int lane = tid & 31;
  const int w    = tid >> 5;
  const int m    = lane & 15;
  const int g    = lane >> 4;

  // async-copy assignment: thread owns 4 consecutive 16B chunks of one row
  const int arow = tid >> 4;
  const int aseg = tid & 15;

  // elementwise assignment: one hidden column per thread, 8 rows each;
  // layernorm gains/shifts for this column hoisted into registers.
  const int hc = tid & 127;
  const int rh = (tid >> 7) * 8;
  const float gI = gains[hc],            sI = shifts[hc];
  const float gJ = gains[H_ + hc],       sJ = shifts[H_ + hc];
  const float gF = gains[2 * H_ + hc],   sF = shifts[2 * H_ + hc];
  const float gO = gains[3 * H_ + hc],   sO = shifts[3 * H_ + hc];
  const float gC = gains[4 * H_ + hc],   sCsh = shifts[4 * H_ + hc];
  // sHP scatter coords for this hc
  const int skc   = hc >> 5;
  const int sgg   = (hc >> 3) & 1;
  const int sfi   = (hc & 7) | (((hc >> 4) & 1) << 3);

  for (int t = 0; t < T_; ++t) {
    // ---- issue async global->LDS copy of xz tile for step t ----
    {
      const size_t gbase = ((size_t)(bbase + arow) * T_ + t) * FOURH;
      #pragma unroll
      for (int q = 0; q < 4; ++q) {
        int hoff = (aseg * 4 + q) * 8;                       // 8 halfs = 16 B
        unsigned laddr = (unsigned)(size_t)&sX[arow * FOURH + hoff];
        unsigned long long gaddr = (unsigned long long)(size_t)(xz + gbase + hoff);
        asm volatile("global_load_async_to_lds_b128 %0, %1, off"
                     :: "v"(laddr), "v"(gaddr) : "memory");
      }
      if (t + 1 < T_) __builtin_prefetch(xz + gbase + FOURH, 0, 0);  // next step -> L2
    }

    // ---- h @ Wh with WMMA (overlaps the async copy) ----
    v8f acc[4] = {};
    #pragma unroll
    for (int kc = 0; kc < KCH; ++kc) {
      v16h a = *(const v16h*)&sHP[(kc * 32 + lane) * 16];
      #pragma unroll
      for (int ct = 0; ct < 4; ++ct) {
        const int cti = w * 4 + ct;
        v16h b = *(const v16h*)&WhP[(((size_t)cti * KCH + kc) * 32 + lane) * 16];
        acc[ct] = __builtin_amdgcn_wmma_f32_16x16x32_f16(false, a, false, b, (short)0, acc[ct], false, false);
      }
    }

    // ---- land async data, fuse xz add into sZ store ----
    asm volatile("s_wait_asynccnt 0x0" ::: "memory");
    __syncthreads();
    #pragma unroll
    for (int ct = 0; ct < 4; ++ct) {
      const int n0 = (w * 4 + ct) * 16;
      #pragma unroll
      for (int r = 0; r < 8; ++r) {
        int row = r + 8 * g;
        sZ[row * FOURH + n0 + m] = acc[ct][r] + (float)sX[row * FOURH + n0 + m];
      }
    }
    __syncthreads();

    // ---- gate layernorm stats: 64 (row,gate) groups, 4 lanes each ----
    {
      const int rg = tid >> 2, row = rg >> 2, gate = rg & 3, sub = tid & 3;
      const float* zr = &sZ[row * FOURH + gate * H_ + sub * 32];
      float s = 0.0f, ss = 0.0f;
      #pragma unroll
      for (int k2 = 0; k2 < 32; ++k2) { float v = zr[k2]; s += v; ss += v * v; }
      s += __shfl_xor(s, 1, 32); ss += __shfl_xor(ss, 1, 32);
      s += __shfl_xor(s, 2, 32); ss += __shfl_xor(ss, 2, 32);
      if (sub == 0) {
        float mean = s * (1.0f / H_);
        float var  = ss * (1.0f / H_) - mean * mean;
        sMean[row][gate] = mean;
        sRstd[row][gate] = rsqrtf(var + 1e-12f);
      }
    }
    __syncthreads();

    // ---- cell update: c = c*sig(f_ln+1) + sig(i_ln)*relu(j_ln) ----
    #pragma unroll
    for (int e = 0; e < 8; ++e) {
      int row = rh + e;
      float iv = sZ[row * FOURH + hc];
      float jv = sZ[row * FOURH + H_ + hc];
      float fv = sZ[row * FOURH + 2 * H_ + hc];
      float iln = (iv - sMean[row][0]) * sRstd[row][0] * gI + sI;
      float jln = (jv - sMean[row][1]) * sRstd[row][1] * gJ + sJ;
      float fln = (fv - sMean[row][2]) * sRstd[row][2] * gF + sF;
      float cn = sC[row * H_ + hc] * sigmoidf_(fln + 1.0f) + sigmoidf_(iln) * fmaxf(jln, 0.0f);
      sC[row * H_ + hc] = cn;
    }
    __syncthreads();

    // ---- cell layernorm stats: 16 rows, 16 lanes each ----
    {
      const int row = tid >> 4, sub = tid & 15;
      const float* cr = &sC[row * H_ + sub * 8];
      float s = 0.0f, ss = 0.0f;
      #pragma unroll
      for (int k2 = 0; k2 < 8; ++k2) { float v = cr[k2]; s += v; ss += v * v; }
      #pragma unroll
      for (int msk = 1; msk < 16; msk <<= 1) {
        s += __shfl_xor(s, msk, 32); ss += __shfl_xor(ss, msk, 32);
      }
      if (sub == 0) {
        float mean = s * (1.0f / H_);
        float var  = ss * (1.0f / H_) - mean * mean;
        sCm[row] = mean;
        sCr[row] = rsqrtf(var + 1e-12f);
      }
    }
    __syncthreads();

    // ---- h = relu(ln(c)) * sig(o_ln); coalesced out + refresh sHP ----
    #pragma unroll
    for (int e = 0; e < 8; ++e) {
      int row = rh + e;
      float cn  = sC[row * H_ + hc];
      float cln = (cn - sCm[row]) * sCr[row] * gC + sCsh;
      float ov  = sZ[row * FOURH + 3 * H_ + hc];
      float oln = (ov - sMean[row][3]) * sRstd[row][3] * gO + sO;
      float hv  = fmaxf(cln, 0.0f) * sigmoidf_(oln);
      out[((size_t)(bbase + row) * T_ + t) * H_ + hc] = hv;
      sHP[(skc * 32 + (row + 16 * sgg)) * 16 + sfi] = (_Float16)hv;
    }
    __syncthreads();
  }
}

// ---------------------------------------------------------------------------
extern "C" void kernel_launch(void* const* d_in, const int* in_sizes, int n_in,
                              void* d_out, int out_size, void* d_ws, size_t ws_size,
                              hipStream_t stream) {
  (void)in_sizes; (void)n_in; (void)out_size; (void)ws_size;
  const int*   x      = (const int*)d_in[0];
  const float* embed  = (const float*)d_in[1];
  const float* W      = (const float*)d_in[2];
  const float* gains  = (const float*)d_in[3];
  const float* shifts = (const float*)d_in[4];
  float* out = (float*)d_out;

  // workspace (f16): WxP[163840] | WhP[65536] | xz[B*T*512]
  _Float16* WxP = (_Float16*)d_ws;
  _Float16* WhP = WxP + (size_t)32 * KCX * 32 * 16;
  _Float16* xz  = WhP + (size_t)32 * KCH * 32 * 16;

  const int cvtTotal = 32 * KCX * 32 * 16 + 32 * KCH * 32 * 16;
  lstm_wprep<<<dim3((cvtTotal + 255) / 256), dim3(256), 0, stream>>>(W, WxP, WhP);
  lstm_xgemm<<<dim3((B_ * T_) / 64, FOURH / 64), dim3(256), 0, stream>>>(x, embed, WxP, xz);
  lstm_scan<<<dim3(B_ / 16), dim3(256), 0, stream>>>(WhP, xz, gains, shifts, out);
}
